// SRUCell_62732292325860
// MI455X (gfx1250) — matrix operand order repaired and checked by reading, not compiled, not run on hardware
//
#include <hip/hip_runtime.h>
#include <hip/hip_bf16.h>
#include <stddef.h>

// ---------------------------------------------------------------------------
// SRU on MI455X (gfx1250)
//   Phase 0: pack x and W once into split-bf16 (hi/lo) arrays laid out in the
//            exact per-lane WMMA fragment order (32 contiguous bytes / lane).
//   Phase 1: u = x @ W via bf16x3 V_WMMA_F32_16X16X32_BF16; inner loop is
//            fragment loads + WMMA only (no conversions).
//   Phase 2: sequential scan over L with deep software prefetch.
// ---------------------------------------------------------------------------

typedef __attribute__((ext_vector_type(16))) __bf16 v16bf;
typedef __attribute__((ext_vector_type(8)))  float  v8f;
typedef __attribute__((ext_vector_type(4)))  float  v4f;

#define LDIM 1024
#define BDIM 32
#define DDIM 1024
#define MDIM (LDIM * BDIM)       // 32768
#define KDIM DDIM                // 1024
#define NDIM (3 * DDIM)          // 3072
#define PLANE ((size_t)MDIM * DDIM)
#define KT    (KDIM / 32)        // 32 k-tiles
#define RTT   (MDIM / 16)        // 2048 row tiles (A)
#define CTT   (NDIM / 16)        // 192 col tiles (W)

// ---------------------- pack x into WMMA A-fragment order -------------------
// A 16x32 bf16 fragment: lane = {M = lane&15, kb = 8*(lane>>4)},
// comp c -> K = kb + (c&7) + 16*(c>>3)  (two contiguous 8-float runs).
__global__ __launch_bounds__(256) void pack_a(const float* __restrict__ A,
                                              __bf16* __restrict__ Ah,
                                              __bf16* __restrict__ Al) {
  const int tile = blockIdx.x * 8 + (threadIdx.x >> 5);  // rt*KT + kt
  const int lane = threadIdx.x & 31;
  const int rt = tile >> 5;        // /KT
  const int kt = tile & 31;
  const float* ap = A + (size_t)(rt * 16 + (lane & 15)) * KDIM
                      + kt * 32 + (lane >> 4) * 8;
  v4f f0 = *(const v4f*)(ap);
  v4f f1 = *(const v4f*)(ap + 4);
  v4f f2 = *(const v4f*)(ap + 16);
  v4f f3 = *(const v4f*)(ap + 20);
  v16bf h, l;
#pragma unroll
  for (int c = 0; c < 4; ++c) {
    float v; __bf16 t;
    v = f0[c]; t = (__bf16)v; h[c]      = t; l[c]      = (__bf16)(v - (float)t);
    v = f1[c]; t = (__bf16)v; h[c + 4]  = t; l[c + 4]  = (__bf16)(v - (float)t);
    v = f2[c]; t = (__bf16)v; h[c + 8]  = t; l[c + 8]  = (__bf16)(v - (float)t);
    v = f3[c]; t = (__bf16)v; h[c + 12] = t; l[c + 12] = (__bf16)(v - (float)t);
  }
  const size_t o = ((size_t)tile * 32 + lane);           // v16bf index
  ((v16bf*)Ah)[o] = h;
  ((v16bf*)Al)[o] = l;
}

// ---------------------- pack W into WMMA B-fragment order -------------------
// B 32x16 bf16 fragment: lane = {N = lane&15, half}, comp c -> K = 8*half + (c&7) + 16*(c>>3).
__global__ __launch_bounds__(256) void pack_w(const float* __restrict__ W,
                                              __bf16* __restrict__ Wh,
                                              __bf16* __restrict__ Wl) {
  const int tile = blockIdx.x * 8 + (threadIdx.x >> 5);  // ct*KT + kt
  const int lane = threadIdx.x & 31;
  const int ct = tile >> 5;
  const int kt = tile & 31;
  const float* wp = W + (size_t)(kt * 32 + (lane >> 4) * 8) * NDIM
                      + ct * 16 + (lane & 15);
  v16bf h, l;
#pragma unroll
  for (int c = 0; c < 16; ++c) {
    float v  = wp[(size_t)((c & 7) + 16 * (c >> 3)) * NDIM];
    __bf16 t = (__bf16)v;
    h[c] = t;
    l[c] = (__bf16)(v - (float)t);
  }
  const size_t o = ((size_t)tile * 32 + lane);
  ((v16bf*)Wh)[o] = h;
  ((v16bf*)Wl)[o] = l;
}

// -------------------------------- GEMM -------------------------------------
__global__ __launch_bounds__(256) void sru_gemm(const __bf16* __restrict__ Ahp,
                                                const __bf16* __restrict__ Alp,
                                                const __bf16* __restrict__ Whp,
                                                const __bf16* __restrict__ Wlp,
                                                float* __restrict__ U) {
  const int lane = threadIdx.x & 31;
  const int wave = threadIdx.x >> 5;        // 0..7
  const int wm   = wave >> 2;               // 0..1
  const int wn   = wave & 3;                // 0..3
  const int rt0  = blockIdx.y * 8 + wm * 4; // first of 4 row tiles
  const int ct0  = blockIdx.x * 8 + wn * 2; // first of 2 col tiles
  const int mloc = lane & 15;
  const int half = lane >> 4;

  const v16bf* Ah = (const v16bf*)Ahp;
  const v16bf* Al = (const v16bf*)Alp;
  const v16bf* Wh = (const v16bf*)Whp;
  const v16bf* Wl = (const v16bf*)Wlp;

  v8f acc[4][2];
#pragma unroll
  for (int mi = 0; mi < 4; ++mi)
#pragma unroll
    for (int ni = 0; ni < 2; ++ni) acc[mi][ni] = (v8f)0.0f;

  for (int kt = 0; kt < KT; ++kt) {
    v16bf ah[4], al[4], bh[2], bl[2];
#pragma unroll
    for (int ni = 0; ni < 2; ++ni) {
      const size_t o = ((size_t)(ct0 + ni) * KT + kt) * 32 + lane;
      bh[ni] = Wh[o];
      bl[ni] = Wl[o];
    }
#pragma unroll
    for (int mi = 0; mi < 4; ++mi) {
      const size_t o = ((size_t)(rt0 + mi) * KT + kt) * 32 + lane;
      ah[mi] = Ah[o];
      al[mi] = Al[o];
    }
#pragma unroll
    for (int mi = 0; mi < 4; ++mi)
#pragma unroll
      for (int ni = 0; ni < 2; ++ni) {
        acc[mi][ni] = __builtin_amdgcn_wmma_f32_16x16x32_bf16(
            false, al[mi], false, bh[ni], (short)0, acc[mi][ni], false, false);
        acc[mi][ni] = __builtin_amdgcn_wmma_f32_16x16x32_bf16(
            false, ah[mi], false, bl[ni], (short)0, acc[mi][ni], false, false);
        acc[mi][ni] = __builtin_amdgcn_wmma_f32_16x16x32_bf16(
            false, ah[mi], false, bh[ni], (short)0, acc[mi][ni], false, false);
      }
  }

  // store: de-interleave column (3d+j) into plane j (coalesced for the scan).
  // C/D layout: comp v -> row = 16*(rt0+mi) + 8*half + v, col = 16*(ct0+ni)+mloc
#pragma unroll
  for (int mi = 0; mi < 4; ++mi)
#pragma unroll
    for (int ni = 0; ni < 2; ++ni) {
      unsigned col = (unsigned)((ct0 + ni) * 16 + mloc);
      unsigned d   = col / 3u;
      unsigned j   = col - 3u * d;
      float* up = U + (size_t)j * PLANE + d;
#pragma unroll
      for (int v = 0; v < 8; ++v) {
        int row = (rt0 + mi) * 16 + half * 8 + v;
        up[(size_t)row * DDIM] = acc[mi][ni][v];
      }
    }
}

// -------------------------------- scan -------------------------------------
__global__ __launch_bounds__(256) void sru_scan(const float* __restrict__ U,
                                                const float* __restrict__ x,
                                                const float* __restrict__ wc,
                                                const float* __restrict__ bias,
                                                const float* __restrict__ c0,
                                                float* __restrict__ out) {
  const int idx = blockIdx.x * 256 + threadIdx.x;   // b*DDIM + d
  const int d = idx & (DDIM - 1);
  const float fw = wc[d];
  const float rw = wc[DDIM + d];
  const float fb = bias[d];
  const float rb = bias[DDIM + d];
  float c = c0[idx];

  const float* u0p = U;
  const float* u1p = U + PLANE;
  const float* u2p = U + 2 * PLANE;

  constexpr int PF = 8;   // deep prefetch ring: only 1 wave/SIMD of occupancy
  float p0[PF], p1[PF], p2[PF], px[PF];
#pragma unroll
  for (int i = 0; i < PF; ++i) {
    size_t o = (size_t)i * MDIM + idx;
    p0[i] = u0p[o]; p1[i] = u1p[o]; p2[i] = u2p[o]; px[i] = x[o];
  }

  for (int l0 = 0; l0 < LDIM; l0 += PF) {
#pragma unroll
    for (int i = 0; i < PF; ++i) {
      const int l = l0 + i;
      float u0 = p0[i];
      float u1 = p1[i] + fb;
      float u2 = p2[i] + rb;
      float xp = px[i];
      const int lp = l + PF;
      if (lp < LDIM) {
        size_t o = (size_t)lp * MDIM + idx;
        p0[i] = u0p[o]; p1[i] = u1p[o]; p2[i] = u2p[o]; px[i] = x[o];
      }
      float f = 1.0f / (1.0f + __expf(-(u1 + c * fw)));
      float r = 1.0f / (1.0f + __expf(-(u2 + c * rw)));
      c = u0 + (c - u0) * f;
      float h = xp + (c - xp) * r;
      out[(size_t)l * MDIM + idx] = h;
    }
  }
  out[(size_t)LDIM * MDIM + idx] = c;   // c_final appended after h
}

// ------------------------------- launch ------------------------------------
extern "C" void kernel_launch(void* const* d_in, const int* in_sizes, int n_in,
                              void* d_out, int out_size, void* d_ws, size_t ws_size,
                              hipStream_t stream) {
  const float* x    = (const float*)d_in[0];   // (L,B,D)
  const float* w    = (const float*)d_in[1];   // (D, 3D)
  const float* wc   = (const float*)d_in[2];   // (2D,)
  const float* bias = (const float*)d_in[3];   // (2D,)
  const float* c0   = (const float*)d_in[4];   // (B,D)
  float* out = (float*)d_out;

  // workspace carve-up (bytes)
  char* ws = (char*)d_ws;
  float*  U  = (float*)ws;                                   // 384 MB
  __bf16* Ah = (__bf16*)(ws + 3 * PLANE * sizeof(float));    //  64 MB
  __bf16* Al = Ah + (size_t)MDIM * KDIM;                     //  64 MB
  __bf16* Wh = Al + (size_t)MDIM * KDIM;                     //   6 MB
  __bf16* Wl = Wh + (size_t)KDIM * NDIM;                     //   6 MB

  pack_a<<<(RTT * KT) / 8, 256, 0, stream>>>(x, Ah, Al);     // 65536 tiles
  pack_w<<<(CTT * KT) / 8, 256, 0, stream>>>(w, Wh, Wl);     //  6144 tiles
  dim3 ggrid(NDIM / 128, MDIM / 128);                        // (24, 256)
  sru_gemm<<<ggrid, 256, 0, stream>>>(Ah, Al, Wh, Wl, U);
  sru_scan<<<MDIM / 256, 256, 0, stream>>>(U, x, wc, bias, c0, out);
}
